// GCNNet_48473000902802
// MI455X (gfx1250) — matrix operand order repaired and checked by reading, not compiled, not run on hardware
//
#include <hip/hip_runtime.h>
#include <hip/hip_bf16.h>

typedef __attribute__((ext_vector_type(16))) __bf16 v16bf;
typedef __attribute__((ext_vector_type(8)))  __bf16 bf16x8;
typedef __attribute__((ext_vector_type(8)))  float  v8f;
typedef __attribute__((ext_vector_type(4)))  unsigned int ui32x4;
typedef __attribute__((ext_vector_type(8)))  int i32x8;
typedef __attribute__((ext_vector_type(4)))  int i32x4;

#define NNODES 65536
#define NGRAPH 512
#define NEDGE  1048576

// ---------------------------------------------------------------- utilities
__global__ __launch_bounds__(256) void k_init(float* deg, int* cnt, int* cursor, int n) {
  int i = blockIdx.x * 256 + threadIdx.x;
  if (i < n) deg[i] = 1.0f;               // self-loop contributes 1 to degree
  if (i < NGRAPH) { cnt[i] = 0; cursor[i] = 0; }
}

__global__ __launch_bounds__(256) void k_edge_prep(const int* __restrict__ src,
                                                   const int* __restrict__ dst,
                                                   float* deg, int* cnt, int e_total) {
  int e = blockIdx.x * 256 + threadIdx.x;
  if (e < e_total) {
    atomicAdd(&deg[dst[e]], 1.0f);
    atomicAdd(&cnt[src[e] >> 7], 1);      // graph id = node / 128
  }
}

__global__ __launch_bounds__(256) void k_dis(float* degdis, int n) {
  int i = blockIdx.x * 256 + threadIdx.x;
  if (i < n) degdis[i] = rsqrtf(degdis[i]);   // deg >= 1 always
}

__global__ __launch_bounds__(512) void k_scan(const int* __restrict__ cnt, int* starts) {
  __shared__ int tmp[NGRAPH];
  int t = threadIdx.x;
  tmp[t] = cnt[t];
  __syncthreads();
  for (int off = 1; off < NGRAPH; off <<= 1) {
    int add = (t >= off) ? tmp[t - off] : 0;
    __syncthreads();
    tmp[t] += add;
    __syncthreads();
  }
  starts[t] = tmp[t] - cnt[t];            // exclusive prefix sum
}

__global__ __launch_bounds__(256) void k_scatter(const int* __restrict__ src,
                                                 int* cursor, const int* __restrict__ starts,
                                                 int* perm, int e_total) {
  int e = blockIdx.x * 256 + threadIdx.x;
  if (e < e_total) {
    int g = src[e] >> 7;
    int pos = atomicAdd(&cursor[g], 1);
    perm[starts[g] + pos] = e;
  }
}

__global__ __launch_bounds__(256) void k_cvt(const float* __restrict__ in,
                                             __bf16* __restrict__ out, int n) {
  int i = blockIdx.x * 256 + threadIdx.x;
  if (i < n) out[i] = (__bf16)in[i];
}

// transpose + convert weights: W[K x Nc] row-major -> Wt[Nc x K] bf16
__global__ __launch_bounds__(256) void k_wt(const float* __restrict__ W,
                                            __bf16* __restrict__ Wt, int K, int Nc) {
  int i = blockIdx.x * 256 + threadIdx.x;
  if (i < K * Nc) {
    int k = i / Nc, nn = i % Nc;
    Wt[(size_t)nn * K + k] = (__bf16)W[i];
  }
}

// ---------------------------------------------------------------- WMMA GEMM
// A: M x K bf16 row-major.  Bt: Nc x K bf16 (B transposed).  C = A*B (+bias)(relu)
// Fragment layouts per CDNA5 ISA (wave32):
//   A 16x32:  lanes 0-15 row M=l, K {0..7,16..23}; lanes 16-31 row M=l-16, K {8..15,24..31}
//   B 32x16:  lanes 0-15 col N=l, K {0..7,16..23}; lanes 16-31 col N=l-16, K {8..15,24..31}
//   C 16x16:  vgpr v, lane l -> row (l>>4)*8 + v, col l&15
__device__ __forceinline__ v16bf load_a_frag(const __bf16* __restrict__ A, int K,
                                             int m0, int k0, int lane) {
  int r  = m0 + (lane & 15);
  int kh = (lane >> 4) * 8;
  const __bf16* p = A + (size_t)r * K + k0 + kh;
  bf16x8 lo = *(const bf16x8*)(p);
  bf16x8 hi = *(const bf16x8*)(p + 16);
  return __builtin_shufflevector(lo, hi, 0,1,2,3,4,5,6,7,8,9,10,11,12,13,14,15);
}

__device__ __forceinline__ v16bf load_b_frag(const __bf16* __restrict__ Bt, int K,
                                             int n0, int k0, int lane) {
  int c  = n0 + (lane & 15);
  int kh = (lane >> 4) * 8;
  const __bf16* p = Bt + (size_t)c * K + k0 + kh;
  bf16x8 lo = *(const bf16x8*)(p);
  bf16x8 hi = *(const bf16x8*)(p + 16);
  return __builtin_shufflevector(lo, hi, 0,1,2,3,4,5,6,7,8,9,10,11,12,13,14,15);
}

template<int NT>
__global__ __launch_bounds__(256)
void k_gemm_bf16(const __bf16* __restrict__ A, const __bf16* __restrict__ Bt,
                 const float* __restrict__ bias,
                 float* __restrict__ outF, __bf16* __restrict__ outH,
                 int M, int K, int Nc, int relu) {
  int lane = threadIdx.x & 31;
  int wave = threadIdx.x >> 5;
  int m0 = blockIdx.x * 128 + wave * 16;
  int n0 = blockIdx.y * (NT * 16);

  v8f acc[NT] = {};

  for (int k0 = 0; k0 < K; k0 += 32) {
    // batch ALL fragment loads for this K-step -> one VMEM clause
    v16bf a = load_a_frag(A, K, m0, k0, lane);
    v16bf b[NT];
#pragma unroll
    for (int t = 0; t < NT; t++)
      b[t] = load_b_frag(Bt, K, n0 + t * 16, k0, lane);
    if (k0 + 32 < K)  // pull next A strip toward the caches
      __builtin_prefetch(A + (size_t)(m0 + (lane & 15)) * K + k0 + 32, 0, 1);
#if __has_builtin(__builtin_amdgcn_sched_barrier)
    // keep the scheduler from sinking these loads back next to their WMMAs:
    // loads stay grouped above, WMMAs retire on partial s_wait_loadcnt values
    __builtin_amdgcn_sched_barrier(0);
#endif
#pragma unroll
    for (int t = 0; t < NT; t++)
      acc[t] = __builtin_amdgcn_wmma_f32_16x16x32_bf16(
          false, a, false, b[t], (short)0, acc[t], false, false);
  }

  int col0  = n0 + (lane & 15);
  int rbase = m0 + (lane >> 4) * 8;
#pragma unroll
  for (int t = 0; t < NT; t++) {
    int c = col0 + t * 16;
    float bv = bias ? bias[c] : 0.0f;
#pragma unroll
    for (int v = 0; v < 8; v++) {
      float val = acc[t][v] + bv;
      if (relu) val = fmaxf(val, 0.0f);
      size_t o = (size_t)(rbase + v) * Nc + c;
      if (outF) outF[o] = val;
      if (outH) outH[o] = (__bf16)val;
    }
  }
}

// ------------------------------------------------- per-graph LDS aggregation
// Block (g, fc): graph g (nodes g*128..+127), feature chunk fc (64 feats).
// acc[i][f] = sum_{e: dst=i} xw[src(e)][f]*dis[src]*dis[dst] + xw[i][f]*dis[i]^2
// H = relu(acc + bias)  (bf16 out)
__global__ __launch_bounds__(256)
void k_agg(const __bf16* __restrict__ XW, const float* __restrict__ dis,
           const int* __restrict__ src, const int* __restrict__ dst,
           const int* __restrict__ perm, const int* __restrict__ starts,
           const int* __restrict__ cnt, const float* __restrict__ bias,
           __bf16* __restrict__ H, int Fout) {
  __shared__ float  accS[128 * 64];
  __shared__ __bf16 xwS [128 * 64];
  int g = blockIdx.x;
  int f0 = blockIdx.y * 64;
  int node0 = g * 128;

#if __has_builtin(__builtin_amdgcn_tensor_load_to_lds)
  // Tensor Data Mover: DMA the 128x64 bf16 tile (row stride Fout) into LDS.
  // D# group0: count=1 | lds_addr | global_addr | type=2 ("image")
  // D# group1: data_size=1(2B), tensor_dim0=Fout, tensor_dim1=128,
  //            tile_dim0=64, tile_dim1=128, tensor_dim0_stride=Fout
  if (threadIdx.x < 32) {
    unsigned long long ga =
        (unsigned long long)(const void*)(XW + (size_t)node0 * Fout + f0);
    unsigned int la = (unsigned int)(unsigned long long)(void*)xwS;
    ui32x4 g0;
    g0[0] = 1u;                                          // count = 1
    g0[1] = la;                                          // lds_addr
    g0[2] = (unsigned int)(ga & 0xffffffffu);            // global_addr[31:0]
    g0[3] = (unsigned int)((ga >> 32) & 0x01ffffffu) | (2u << 30); // addr hi | type=2
    i32x8 g1;
    g1[0] = (1 << 16);                                   // data_size=1 (2 bytes)
    g1[1] = (int)((Fout & 0xffff) << 16);                // tensor_dim0[15:0] @ [63:48]
    g1[2] = (int)(((unsigned)Fout >> 16) | (128u << 16));// dim0 hi | tensor_dim1=128
    g1[3] = (int)(64u << 16);                            // dim1 hi=0 | tile_dim0=64
    g1[4] = 128;                                         // tile_dim1=128, tile_dim2=0
    g1[5] = Fout;                                        // tensor_dim0_stride[31:0]
    g1[6] = 0;                                           // stride hi | dim1_stride lo
    g1[7] = 0;
    i32x4 z4 = {0, 0, 0, 0};
    i32x8 z8 = {0, 0, 0, 0, 0, 0, 0, 0};
    // 6-arg form (clang-23 / therock headers): groups 2,3 unused for 2D tensor
    __builtin_amdgcn_tensor_load_to_lds(g0, g1, z4, z4, z8, 0);
    __builtin_amdgcn_s_wait_tensorcnt(0);
  }
  __syncthreads();
#else
  for (int i = threadIdx.x; i < 128 * 64; i += 256) {
    int r = i >> 6, f = i & 63;
    xwS[i] = XW[(size_t)(node0 + r) * Fout + f0 + f];
  }
  __syncthreads();
#endif

  for (int i = threadIdx.x; i < 128 * 64; i += 256) {
    int r = i >> 6;
    float dv = dis[node0 + r];
    accS[i] = (float)xwS[i] * dv * dv;    // self-loop term
  }
  __syncthreads();

  int e0 = starts[g], ne = cnt[g];
  int f  = threadIdx.x & 63;
  int eg = threadIdx.x >> 6;              // 4 edges in flight per round
  for (int eb = eg; eb < ne; eb += 4) {
    int e = perm[e0 + eb];
    int s = src[e], d = dst[e];
    float nrm = dis[s] * dis[d];
    atomicAdd(&accS[(d & 127) * 64 + f], (float)xwS[(s & 127) * 64 + f] * nrm);
  }
  __syncthreads();

  for (int i = threadIdx.x; i < 128 * 64; i += 256) {
    int r = i >> 6, ff = i & 63;
    float v = fmaxf(accS[i] + bias[f0 + ff], 0.0f);
    H[(size_t)(node0 + r) * Fout + f0 + ff] = (__bf16)v;
  }
}

// ------------------------------------------------------------- segment max
__global__ __launch_bounds__(256)
void k_segmax(const __bf16* __restrict__ H, __bf16* __restrict__ P, int Fout) {
  int g = blockIdx.x, f = threadIdx.x;
  const __bf16* p = H + (size_t)g * 128 * Fout + f;
  float m = -3.4e38f;
  for (int r = 0; r < 128; r++) m = fmaxf(m, (float)p[(size_t)r * Fout]);
  P[g * Fout + f] = (__bf16)m;
}

// ---------------------------------------------------------------- launcher
extern "C" void kernel_launch(void* const* d_in, const int* in_sizes, int n_in,
                              void* d_out, int out_size, void* d_ws, size_t ws_size,
                              hipStream_t stream) {
  const float* x   = (const float*)d_in[0];
  const int*   src = (const int*)  d_in[1];
  const int*   dst = (const int*)  d_in[2];
  const float* W1  = (const float*)d_in[4];
  const float* b1  = (const float*)d_in[5];
  const float* W2  = (const float*)d_in[6];
  const float* b2  = (const float*)d_in[7];
  const float* W3  = (const float*)d_in[8];
  const float* b3  = (const float*)d_in[9];
  const float* fW1 = (const float*)d_in[10];
  const float* fb1 = (const float*)d_in[11];
  const float* fW2 = (const float*)d_in[12];
  const float* fb2 = (const float*)d_in[13];

  const int N = NNODES, E = NEDGE;
  char* ws = (char*)d_ws;
  size_t off = 0;
  auto alloc = [&](size_t bytes) -> void* {
    void* p = ws + off;
    off += (bytes + 255) & ~(size_t)255;
    return p;
  };

  float*  degdis = (float*) alloc((size_t)N * 4);
  int*    cnt    = (int*)   alloc(NGRAPH * 4);
  int*    starts = (int*)   alloc(NGRAPH * 4);
  int*    cursor = (int*)   alloc(NGRAPH * 4);
  int*    perm   = (int*)   alloc((size_t)E * 4);
  __bf16* W1t    = (__bf16*)alloc((size_t)64  * 64   * 2);
  __bf16* W2t    = (__bf16*)alloc((size_t)128 * 64   * 2);
  __bf16* W3t    = (__bf16*)alloc((size_t)256 * 128  * 2);
  __bf16* fW1t   = (__bf16*)alloc((size_t)1024* 256  * 2);
  __bf16* fW2t   = (__bf16*)alloc((size_t)128 * 1024 * 2);
  __bf16* bufH   = (__bf16*)alloc((size_t)N * 256 * 2);   // current features
  __bf16* bufXW  = (__bf16*)alloc((size_t)N * 256 * 2);   // X*W pre-aggregation
  __bf16* bufG1  = (__bf16*)alloc((size_t)N * 1024 * 2);  // dense-head hidden
  __bf16* pmax   = (__bf16*)alloc((size_t)NGRAPH * 256 * 2);
  __bf16* p1     = (__bf16*)alloc((size_t)NGRAPH * 1024 * 2);

  float* outDense = (float*)d_out;                 // [65536,128]
  float* outPool  = (float*)d_out + (size_t)N * 128;

  // ---- graph structure (once, reused by all 3 layers)
  k_init     <<<(N + 255) / 256, 256, 0, stream>>>(degdis, cnt, cursor, N);
  k_edge_prep<<<E / 256,        256, 0, stream>>>(src, dst, degdis, cnt, E);
  k_dis      <<<N / 256,        256, 0, stream>>>(degdis, N);
  k_scan     <<<1,              512, 0, stream>>>(cnt, starts);
  k_scatter  <<<E / 256,        256, 0, stream>>>(src, cursor, starts, perm, E);

  // ---- operand conversion
  k_cvt<<<(N * 64) / 256, 256, 0, stream>>>(x, bufH, N * 64);
  k_wt <<<(64 * 64   + 255) / 256, 256, 0, stream>>>(W1,  W1t,  64,   64);
  k_wt <<<(64 * 128  + 255) / 256, 256, 0, stream>>>(W2,  W2t,  64,   128);
  k_wt <<<(128 * 256 + 255) / 256, 256, 0, stream>>>(W3,  W3t,  128,  256);
  k_wt <<<(256 * 1024+ 255) / 256, 256, 0, stream>>>(fW1, fW1t, 256,  1024);
  k_wt <<<(1024* 128 + 255) / 256, 256, 0, stream>>>(fW2, fW2t, 1024, 128);

  // ---- GCN layer 1: [N,64] -> [N,64]
  k_gemm_bf16<4><<<dim3(N / 128, 1), 256, 0, stream>>>(bufH, W1t, nullptr, nullptr, bufXW, N, 64, 64, 0);
  k_agg<<<dim3(NGRAPH, 1), 256, 0, stream>>>(bufXW, degdis, src, dst, perm, starts, cnt, b1, bufH, 64);

  // ---- GCN layer 2: [N,64] -> [N,128]
  k_gemm_bf16<8><<<dim3(N / 128, 1), 256, 0, stream>>>(bufH, W2t, nullptr, nullptr, bufXW, N, 64, 128, 0);
  k_agg<<<dim3(NGRAPH, 2), 256, 0, stream>>>(bufXW, degdis, src, dst, perm, starts, cnt, b2, bufH, 128);

  // ---- GCN layer 3: [N,128] -> [N,256]
  k_gemm_bf16<8><<<dim3(N / 128, 2), 256, 0, stream>>>(bufH, W3t, nullptr, nullptr, bufXW, N, 128, 256, 0);
  k_agg<<<dim3(NGRAPH, 4), 256, 0, stream>>>(bufXW, degdis, src, dst, perm, starts, cnt, b3, bufH, 256);

  // ---- dense per-node head: relu(H @ fW1 + fb1) @ fW2 + fb2 -> d_out[0:N*128]
  k_gemm_bf16<8><<<dim3(N / 128, 8), 256, 0, stream>>>(bufH,  fW1t, fb1, nullptr, bufG1, N, 256, 1024, 1);
  k_gemm_bf16<8><<<dim3(N / 128, 1), 256, 0, stream>>>(bufG1, fW2t, fb2, outDense, nullptr, N, 1024, 128, 0);

  // ---- global max pool head -> d_out[N*128 : N*128 + 512*128]
  k_segmax<<<NGRAPH, 256, 0, stream>>>(bufH, pmax, 256);
  k_gemm_bf16<8><<<dim3(NGRAPH / 128, 8), 256, 0, stream>>>(pmax, fW1t, fb1, nullptr, p1, NGRAPH, 256, 1024, 1);
  k_gemm_bf16<8><<<dim3(NGRAPH / 128, 1), 256, 0, stream>>>(p1,  fW2t, fb2, outPool, nullptr, NGRAPH, 1024, 128, 0);
}
